// AttnBlock_41532333752703
// MI455X (gfx1250) — compile-verified, matrix operand order checked by reading, not run on hardware
//
#include <hip/hip_runtime.h>

typedef __attribute__((ext_vector_type(16))) _Float16 v16h;
typedef __attribute__((ext_vector_type(8)))  _Float16 v8h;
typedef __attribute__((ext_vector_type(8)))  float    v8f;

union H16 { v16h v; v8h h[2]; _Float16 e[16]; };
union H8  { v8h v; _Float16 e[8]; };

// Fragment for one 16x32 (f16) operand slice. Per ISA 7.12.2:
//   lane in [0,15]:  e[0..7]=K(base+0..7),  e[8..15]=K(base+16..23)
//   lane in [16,31]: base shifted by 8 (caller passes kb = 8*(lane>>4))
// Works for A (row = M) and for B when stored "transposed" [N][K].
__device__ inline v16h frag_h16(const _Float16* row) {
  H16 f;
  f.h[0] = *(const v8h*)(row);
  f.h[1] = *(const v8h*)(row + 16);
  return f.v;
}

__device__ inline v8f wmma16(v16h a, v16h b, v8f c) {
  return __builtin_amdgcn_wmma_f32_16x16x32_f16(false, a, false, b, (short)0, c, false, false);
}

#define HSTRIDE 264   // 256 + 8 f16 pad -> conflict-free b128 row reads

// ---------------- Kernel W: one-shot f32 -> f16 weight conversion ----------------
// dst holds 4 row-major 256x256 f16 matrices: [Wq | Wk | Wv | Wo]
__global__ __launch_bounds__(256) void k_wcvt(const float* __restrict__ wq,
                                              const float* __restrict__ wk,
                                              const float* __restrict__ wv,
                                              const float* __restrict__ wo,
                                              _Float16* __restrict__ dst) {
  int m = blockIdx.x >> 5;                                   // 0..3
  int idx = (((blockIdx.x & 31) * 256) + threadIdx.x) * 8;   // 8 elems / thread
  const float* src = (m == 0) ? wq : (m == 1) ? wk : (m == 2) ? wv : wo;
  float4 a = ((const float4*)(src + idx))[0];
  float4 b = ((const float4*)(src + idx))[1];
  H8 h;
  h.e[0] = (_Float16)a.x; h.e[1] = (_Float16)a.y; h.e[2] = (_Float16)a.z; h.e[3] = (_Float16)a.w;
  h.e[4] = (_Float16)b.x; h.e[5] = (_Float16)b.y; h.e[6] = (_Float16)b.z; h.e[7] = (_Float16)b.w;
  *(v8h*)(dst + (size_t)m * 65536 + idx) = h.v;
}

// ---------------- Kernel A: GroupNorm statistics ----------------
__global__ __launch_bounds__(256) void k_stats(const float* __restrict__ x,
                                               float* __restrict__ stats) {
  int g = blockIdx.x;                 // 0..127 = b*32 + group
  int b = g >> 5, grp = g & 31;
  const float* base = x + ((size_t)(b * 256 + grp * 8)) * 65536;
  float s = 0.f, ss = 0.f;
  for (int c = 0; c < 8; ++c) {
    const float4* p = (const float4*)(base + (size_t)c * 65536);
    for (int i = threadIdx.x; i < 16384; i += 256) {
      float4 v = p[i];
      s  += v.x + v.y + v.z + v.w;
      ss += v.x * v.x + v.y * v.y + v.z * v.z + v.w * v.w;
    }
  }
  __shared__ float rs[256], rss[256];
  rs[threadIdx.x] = s; rss[threadIdx.x] = ss;
  __syncthreads();
  for (int o = 128; o > 0; o >>= 1) {
    if (threadIdx.x < o) { rs[threadIdx.x] += rs[threadIdx.x + o]; rss[threadIdx.x] += rss[threadIdx.x + o]; }
    __syncthreads();
  }
  if (threadIdx.x == 0) {
    const float inv = 1.0f / 524288.0f;
    float mean = rs[0] * inv;
    float var  = rss[0] * inv - mean * mean;
    stats[g * 2]     = mean;
    stats[g * 2 + 1] = rsqrtf(var + 1e-6f);
  }
}

// ---------------- Kernel B: GroupNorm apply + Q/K/V 1x1 convs ----------------
// Outputs: Qt[w][t][c], Kt[w][t][c] (transposed), V[w][c][t]  -- all f16.
__global__ __launch_bounds__(256, 1) void k_qkv(
    const float* __restrict__ x, const float* __restrict__ gamma, const float* __restrict__ beta,
    const _Float16* __restrict__ whq, const float* __restrict__ bq,
    const _Float16* __restrict__ whk, const float* __restrict__ bk,
    const _Float16* __restrict__ whv, const float* __restrict__ bv,
    const float* __restrict__ stats,
    _Float16* __restrict__ Qt, _Float16* __restrict__ Kt, _Float16* __restrict__ V) {
  __shared__ _Float16 hT[256 * HSTRIDE];   // h transposed: [t][c]
  __shared__ float sc[256], sh[256];
  int w = blockIdx.x;
  int b = w >> 8, ph = (w >> 4) & 15, pw = w & 15;
  int tid = threadIdx.x;
  {
    int c = tid;
    int g = b * 32 + (c >> 3);
    float mean = stats[g * 2], rstd = stats[g * 2 + 1];
    float s = rstd * gamma[c];
    sc[c] = s;
    sh[c] = beta[c] - mean * s;
  }
  __syncthreads();
  {
    int t = tid;   // token = 16*r + col
    size_t base = (size_t)b * 256 * 65536 + (size_t)(ph * 16 + (t >> 4)) * 256 + pw * 16 + (t & 15);
    for (int c0 = 0; c0 < 256; c0 += 8) {
      H8 hv;
#pragma unroll
      for (int i = 0; i < 8; ++i) {
        float xv = x[base + (size_t)(c0 + i) * 65536];
        hv.e[i] = (_Float16)(xv * sc[c0 + i] + sh[c0 + i]);
      }
      *(v8h*)(&hT[t * HSTRIDE + c0]) = hv.v;
    }
  }
  __syncthreads();
  int lane = tid & 31, wv_ = tid >> 5;
  int r = lane & 15, hh = lane >> 4, kb = hh * 8;
  size_t wbase = (size_t)w * 65536;
  for (int job = wv_; job < 48; job += 8) {
    if (job < 32) {
      // Q/K: D[cout][t] = W.h, stored transposed -> [t][cout]
      int mat = job >> 4;
      int ct  = job & 15;
      const _Float16* W = mat ? whk : whq;
      const float* B = mat ? bk : bq;
      _Float16* Out  = (mat ? Kt : Qt) + wbase;
      v16h wa[8];
#pragma unroll
      for (int kc = 0; kc < 8; ++kc)
        wa[kc] = frag_h16(W + (ct * 16 + r) * 256 + kc * 32 + kb);
      float bias[8];
#pragma unroll
      for (int j = 0; j < 8; ++j) bias[j] = B[ct * 16 + j + 8 * hh];
      for (int tt = 0; tt < 16; ++tt) {
        v8f acc;
#pragma unroll
        for (int j = 0; j < 8; ++j) acc[j] = bias[j];
#pragma unroll
        for (int kc = 0; kc < 8; ++kc) {
          v16h bf = frag_h16(&hT[(tt * 16 + r) * HSTRIDE + kc * 32 + kb]);
          acc = wmma16(wa[kc], bf, acc);
        }
        H8 o;
#pragma unroll
        for (int j = 0; j < 8; ++j) o.e[j] = (_Float16)acc[j];
        // element (cout=ct*16+j+8hh, t=tt*16+r) -> Out[t][cout], contiguous in j
        *(v8h*)(Out + (size_t)(tt * 16 + r) * 256 + ct * 16 + kb) = o.v;
      }
    } else {
      // V^T formulation: D[t][cout] = h^T . Wv^T, stored as V[cout][t]
      int tt = job & 15;
      v16h ha[8];
#pragma unroll
      for (int kc = 0; kc < 8; ++kc)
        ha[kc] = frag_h16(&hT[(tt * 16 + r) * HSTRIDE + kc * 32 + kb]);
      for (int ct = 0; ct < 16; ++ct) {
        float bvv = bv[ct * 16 + r];
        v8f acc;
#pragma unroll
        for (int j = 0; j < 8; ++j) acc[j] = bvv;
#pragma unroll
        for (int kc = 0; kc < 8; ++kc) {
          v16h bf = frag_h16(whv + (ct * 16 + r) * 256 + kc * 32 + kb);
          acc = wmma16(ha[kc], bf, acc);
        }
        H8 o;
#pragma unroll
        for (int j = 0; j < 8; ++j) o.e[j] = (_Float16)acc[j];
        // element (t=tt*16+j+8hh, cout=ct*16+r) -> V[cout][t], contiguous in j
        *(v8h*)(V + wbase + (size_t)(ct * 16 + r) * 256 + tt * 16 + kb) = o.v;
      }
    }
  }
}

// ---------------- Kernel C: windowed attention ----------------
__global__ __launch_bounds__(256, 1) void k_attn(
    const _Float16* __restrict__ Qt, const _Float16* __restrict__ Kt,
    const _Float16* __restrict__ V, _Float16* __restrict__ O) {
  __shared__ _Float16 kls[256 * HSTRIDE];     // Kt window: [k-token][c]
  __shared__ _Float16 P[8][16 * HSTRIDE];     // per-wave softmax strip [q][k]
  int w = blockIdx.x;
  size_t wbase = (size_t)w * 65536;
  int tid = threadIdx.x;
  for (int idx = tid; idx < 8192; idx += 256) {
    int row = idx >> 5, cc = (idx & 31) * 8;
    *(v8h*)(&kls[row * HSTRIDE + cc]) = *(const v8h*)(Kt + wbase + (size_t)row * 256 + cc);
  }
  __syncthreads();
  int lane = tid & 31, wv_ = tid >> 5;
  int r = lane & 15, hh = lane >> 4, kb = hh * 8;
  const float scale = 0.0625f;                 // 256^-0.5
  for (int s = wv_; s < 16; s += 8) {
    int q0 = s * 16;
    v16h qa[8];
#pragma unroll
    for (int kc = 0; kc < 8; ++kc)
      qa[kc] = frag_h16(Qt + wbase + (size_t)(q0 + r) * 256 + kc * 32 + kb);
    // S strip: 16 k-tiles kept in registers
    v8f sacc[16];
#pragma unroll
    for (int kt_i = 0; kt_i < 16; ++kt_i) {
      v8f acc = {0.f,0.f,0.f,0.f,0.f,0.f,0.f,0.f};
#pragma unroll
      for (int kc = 0; kc < 8; ++kc) {
        v16h bf = frag_h16(&kls[(kt_i * 16 + r) * HSTRIDE + kc * 32 + kb]);
        acc = wmma16(qa[kc], bf, acc);
      }
      sacc[kt_i] = acc;
    }
    // softmax over k (per q row = (j + 8*hh); row spans 16 lanes of this half)
    float rinv[8];
#pragma unroll
    for (int j = 0; j < 8; ++j) {
      float m = sacc[0][j];
#pragma unroll
      for (int kt_i = 1; kt_i < 16; ++kt_i) m = fmaxf(m, sacc[kt_i][j]);
#pragma unroll
      for (int mk = 1; mk <= 8; mk <<= 1) m = fmaxf(m, __shfl_xor(m, mk, 32));
      float sum = 0.f;
#pragma unroll
      for (int kt_i = 0; kt_i < 16; ++kt_i) {
        float e = __expf((sacc[kt_i][j] - m) * scale);
        sacc[kt_i][j] = e;
        sum += e;
      }
#pragma unroll
      for (int mk = 1; mk <= 8; mk <<= 1) sum += __shfl_xor(sum, mk, 32);
      rinv[j] = 1.0f / sum;
    }
    _Float16* Pb = P[wv_];
#pragma unroll
    for (int kt_i = 0; kt_i < 16; ++kt_i) {
#pragma unroll
      for (int j = 0; j < 8; ++j)
        Pb[(j + 8 * hh) * HSTRIDE + kt_i * 16 + r] = (_Float16)(sacc[kt_i][j] * rinv[j]);
    }
    // O^T = P . V^T  (A = P strip, B[k][c] = V[c][k] read straight from global)
    v16h pa[8];
#pragma unroll
    for (int kc = 0; kc < 8; ++kc)
      pa[kc] = frag_h16(Pb + r * HSTRIDE + kc * 32 + kb);
    for (int ct = 0; ct < 16; ++ct) {
      v8f acc = {0.f,0.f,0.f,0.f,0.f,0.f,0.f,0.f};
#pragma unroll
      for (int kc = 0; kc < 8; ++kc) {
        v16h bf = frag_h16(V + wbase + (size_t)(ct * 16 + r) * 256 + kc * 32 + kb);
        acc = wmma16(pa[kc], bf, acc);
      }
      H8 o;
#pragma unroll
      for (int j = 0; j < 8; ++j) o.e[j] = (_Float16)acc[j];
      // element (q=q0+j+8hh, c=ct*16+r) -> O[c][t], contiguous in j
      *(v8h*)(O + wbase + (size_t)(ct * 16 + r) * 256 + q0 + kb) = o.v;
    }
  }
}

// ---------------- Kernel D: output 1x1 conv + residual ----------------
__global__ __launch_bounds__(256, 1) void k_out(
    const float* __restrict__ x, const _Float16* __restrict__ O,
    const _Float16* __restrict__ who, const float* __restrict__ bo,
    float* __restrict__ out) {
  __shared__ _Float16 oT[256 * HSTRIDE];   // O transposed: [t][c]
  int w = blockIdx.x;
  int b = w >> 8, ph = (w >> 4) & 15, pw = w & 15;
  size_t wbase = (size_t)w * 65536;
  int tid = threadIdx.x;
  for (int idx = tid; idx < 8192; idx += 256) {
    int c = idx >> 5, tc = (idx & 31) * 8;
    H8 v_; v_.v = *(const v8h*)(O + wbase + (size_t)c * 256 + tc);
#pragma unroll
    for (int i = 0; i < 8; ++i) oT[(tc + i) * HSTRIDE + c] = v_.e[i];
  }
  __syncthreads();
  int lane = tid & 31, wv_ = tid >> 5;
  int r = lane & 15, hh = lane >> 4, kb = hh * 8;
  for (int ct = wv_; ct < 16; ct += 8) {
    v16h wa[8];
#pragma unroll
    for (int kc = 0; kc < 8; ++kc)
      wa[kc] = frag_h16(who + (ct * 16 + r) * 256 + kc * 32 + kb);
    float bias[8];
#pragma unroll
    for (int j = 0; j < 8; ++j) bias[j] = bo[ct * 16 + j + 8 * hh];
    for (int tt = 0; tt < 16; ++tt) {
      v8f acc;
#pragma unroll
      for (int j = 0; j < 8; ++j) acc[j] = bias[j];
#pragma unroll
      for (int kc = 0; kc < 8; ++kc) {
        v16h bf = frag_h16(&oT[(tt * 16 + r) * HSTRIDE + kc * 32 + kb]);
        acc = wmma16(wa[kc], bf, acc);
      }
      // element (cout=ct*16+j+8hh, token t=tt*16+r -> pixel row=ph*16+tt, col=pw*16+r)
      size_t base = ((size_t)(b * 256 + ct * 16 + 8 * hh) * 256 + ph * 16 + tt) * 256 + pw * 16 + r;
#pragma unroll
      for (int j = 0; j < 8; ++j) {
        size_t a = base + (size_t)j * 65536;
        out[a] = x[a] + acc[j];
      }
    }
  }
}

extern "C" void kernel_launch(void* const* d_in, const int* in_sizes, int n_in,
                              void* d_out, int out_size, void* d_ws, size_t ws_size,
                              hipStream_t stream) {
  (void)in_sizes; (void)n_in; (void)out_size; (void)ws_size;
  const float* x     = (const float*)d_in[0];
  const float* gamma = (const float*)d_in[1];
  const float* beta  = (const float*)d_in[2];
  const float* wq    = (const float*)d_in[3];
  const float* bq    = (const float*)d_in[4];
  const float* wk    = (const float*)d_in[5];
  const float* bk    = (const float*)d_in[6];
  const float* wv    = (const float*)d_in[7];
  const float* bv    = (const float*)d_in[8];
  const float* wo    = (const float*)d_in[9];
  const float* bo    = (const float*)d_in[10];
  float* out = (float*)d_out;

  char* ws = (char*)d_ws;
  float* stats = (float*)ws;                                        // 1 KB
  _Float16* WH = (_Float16*)(ws + 4096);                            // 4 x 128 KB f16 weights
  const size_t WOFF = 4096 + 4 * (size_t)65536 * sizeof(_Float16);  // = 528384
  const size_t BUF = (size_t)1024 * 65536 * sizeof(_Float16);       // 128 MB each
  _Float16* Qt = (_Float16*)(ws + WOFF);
  _Float16* Kt = (_Float16*)(ws + WOFF + BUF);
  _Float16* V  = (_Float16*)(ws + WOFF + 2 * BUF);
  _Float16* O  = (_Float16*)(ws + WOFF + 3 * BUF);
  _Float16* whq = WH;
  _Float16* whk = WH + 65536;
  _Float16* whv = WH + 2 * 65536;
  _Float16* who = WH + 3 * 65536;

  k_wcvt<<<128, 256, 0, stream>>>(wq, wk, wv, wo, WH);
  k_stats<<<128, 256, 0, stream>>>(x, stats);
  k_qkv<<<1024, 256, 0, stream>>>(x, gamma, beta, whq, bq, whk, bk, whv, bv, stats, Qt, Kt, V);
  k_attn<<<1024, 256, 0, stream>>>(Qt, Kt, V, O);
  k_out<<<1024, 256, 0, stream>>>(x, O, who, bo, out);
}
